// ChebConvGAD_c_65300682768503
// MI455X (gfx1250) — matrix operand order repaired and checked by reading, not compiled, not run on hardware
//
#include <hip/hip_runtime.h>

// ---------- types ----------
typedef __attribute__((ext_vector_type(16))) __bf16 v16bf;
typedef __attribute__((ext_vector_type(8)))  float  v8f;
typedef __attribute__((ext_vector_type(4)))  unsigned int v4u;
typedef __attribute__((ext_vector_type(8)))  int v8i;
typedef __attribute__((ext_vector_type(4)))  int v4i;

struct U4 { unsigned int x, y, z, w; };
union AB16 { U4 u[2]; v16bf v; };

__device__ __forceinline__ unsigned short f2b(float f) {
  unsigned int u = __float_as_uint(f);
  u += 0x7fffu + ((u >> 16) & 1u);            // round-to-nearest-even
  return (unsigned short)(u >> 16);
}
__device__ __forceinline__ float b2f(unsigned int hbits) {
  return __uint_as_float(hbits << 16);
}

// ---------- small utility kernels ----------
__global__ void k_zero(float* __restrict__ p, int n) {
  int i = blockIdx.x * 256 + threadIdx.x;
  if (i < n) p[i] = 0.0f;
}

__global__ void k_cvt(const float* __restrict__ in, unsigned short* __restrict__ out, int n) {
  int i = blockIdx.x * 256 + threadIdx.x;
  if (i < n) out[i] = f2b(in[i]);
}

__global__ void k_deg(const int* __restrict__ ei, float* __restrict__ deg, int E) {
  int e = blockIdx.x * 256 + threadIdx.x;
  if (e < E) atomicAdd(&deg[ei[e]], 1.0f);
}

__global__ void k_rsqrt(const float* __restrict__ deg, float* __restrict__ dis, int n) {
  int i = blockIdx.x * 256 + threadIdx.x;
  if (i < n) {
    float d = deg[i];
    dis[i] = (d > 0.0f) ? rsqrtf(d) : 0.0f;
  }
}

// gather h[src] (bf16), scale by norm, atomic-scatter fp32 into txf[dst]
__global__ void k_scatter(const int* __restrict__ ei, const float* __restrict__ dis,
                          const unsigned short* __restrict__ h, float* __restrict__ txf,
                          int E) {
  int gid = blockIdx.x * 256 + threadIdx.x;
  int e = gid >> 5;
  if (e >= E) return;
  int c = (gid & 31) * 8;
  int src = ei[e];
  int dst = ei[E + e];
  float nrm = -dis[src] * dis[dst];
  U4 d = *((const U4*)h + ((size_t)src * 32 + (c >> 3)));
  float* o = txf + (size_t)dst * 256 + c;
  unsigned int ws[4] = {d.x, d.y, d.z, d.w};
#pragma unroll
  for (int i = 0; i < 4; ++i) {
    atomicAdd(o + 2 * i,     nrm * b2f(ws[i] & 0xffffu));
    atomicAdd(o + 2 * i + 1, nrm * __uint_as_float(ws[i] & 0xffff0000u));
  }
}

// ---------- TDM: DMA a [upto 64]x256 bf16 tile into LDS with 4-DWORD row padding ----------
// LDS row stride becomes 256*2 + 16 = 528 B (= 33 U4), 16B-aligned and bank-spread.
__device__ __forceinline__ void tdm_load_tile(const unsigned short* gsrc, unsigned int ldsByteOff,
                                              int rowsRemaining, int rowStart) {
  unsigned long long ga = (unsigned long long)(size_t)gsrc + (unsigned long long)rowStart * 512ull;
  v4u g0;
  g0[0] = 1u;                                             // count=1, user descriptor
  g0[1] = ldsByteOff;                                     // lds_addr (bytes)
  g0[2] = (unsigned int)(ga & 0xffffffffu);               // global_addr[31:0]
  g0[3] = (unsigned int)((ga >> 32) & 0x1ffffffu)         // global_addr[56:32]
        | (2u << 30);                                     // type = 2 ("image")
  v8i g1;
  g1[0] = (int)((1u << 16)                                // data_size = 2 bytes
              | (1u << 20)                                // pad_enable
              | (6u << 22)                                // pad_interval: 128 DWORDs (one 256-elem row)
              | (3u << 25));                              // pad_amount: 4 DWORDs (16 B)
  g1[1] = (int)(256u << 16);                              // tensor_dim0 = 256 (low 16 in [63:48])
  g1[2] = (int)(((unsigned int)rowsRemaining & 0xffffu) << 16); // dim0 hi=0 | tensor_dim1 lo
  g1[3] = (int)((((unsigned int)rowsRemaining >> 16) & 0xffffu)
              | (256u << 16));                            // tensor_dim1 hi | tile_dim0 = 256
  g1[4] = 64;                                             // tile_dim1 = 64 rows, tile_dim2 = 0
  g1[5] = 256;                                            // tensor_dim0_stride = 256 elements
  g1[6] = 0;
  g1[7] = 0;
  v4i z4 = {0, 0, 0, 0};
  v8i z8 = {0, 0, 0, 0, 0, 0, 0, 0};
  __builtin_amdgcn_tensor_load_to_lds(g0, g1, z4, z4, z8, 0);
}

// ---------- WMMA GEMM: out = [relu](A1@B1^T (+ A2@B2^T) + bias), [N,256]x[256,256] ----------
// block = 256 threads = 8 waves; block computes a 64-row x 256-col output tile.
// wave w owns cols [32w,32w+32): 4 row-tiles x 2 col-tiles = 8 accumulators.
template <bool RELU, bool DUAL>
__global__ __launch_bounds__(256) void k_gemm(
    const unsigned short* __restrict__ A1, const unsigned short* __restrict__ B1,
    const unsigned short* __restrict__ A2, const unsigned short* __restrict__ B2,
    const float* __restrict__ bias, unsigned short* __restrict__ out, int N) {
  __shared__ U4 sA[64 * 33];                    // 64 rows x (32+1 pad) U4 of bf16
  const int tid   = threadIdx.x;
  const int lane  = tid & 31;
  const int wave  = tid >> 5;
  const int n0    = wave * 32;
  const int rrow  = lane & 15;
  const int khalf = lane >> 4;
  const int brow  = blockIdx.x * 64;

  v8f acc[4][2] = {};

  const unsigned int ldsOff = (unsigned int)(size_t)(void*)&sA[0];

#pragma unroll
  for (int pass = 0; pass < (DUAL ? 2 : 1); ++pass) {
    const unsigned short* A = pass ? A2 : A1;
    const unsigned short* B = pass ? B2 : B1;
    if (pass) __syncthreads();                  // previous tile fully consumed
    if (wave == 0) {                            // one TDM per workgroup pass
      tdm_load_tile(A, ldsOff, N - brow, brow);
      __builtin_amdgcn_s_wait_tensorcnt((short)0);
    }
    __syncthreads();

    const U4* Bp = (const U4*)B;
#pragma unroll
    for (int ks = 0; ks < 8; ++ks) {            // K = 8 * 32
      // B fragments (32x16): lane half kh holds K = kh*16..+15 of col n (contig. in W row n)
      AB16 bf0, bf1;
      const size_t jb = (size_t)(n0 + rrow) * 32 + ks * 4 + khalf * 2;
      bf0.u[0] = Bp[jb];       bf0.u[1] = Bp[jb + 1];
      bf1.u[0] = Bp[jb + 512]; bf1.u[1] = Bp[jb + 513];   // +16 weight rows
      const int ja = ks * 4 + khalf * 2;
#pragma unroll
      for (int t = 0; t < 4; ++t) {
        // A fragment (16x32): lane half kh holds K = kh*16..+15
        AB16 af;
        af.u[0] = sA[(t * 16 + rrow) * 33 + ja];
        af.u[1] = sA[(t * 16 + rrow) * 33 + ja + 1];
        acc[t][0] = __builtin_amdgcn_wmma_f32_16x16x32_bf16(false, af.v, false, bf0.v, (short)0, acc[t][0], false, false);
        acc[t][1] = __builtin_amdgcn_wmma_f32_16x16x32_bf16(false, af.v, false, bf1.v, (short)0, acc[t][1], false, false);
      }
    }
  }

  // epilogue: C/D layout -> row = r + 8*khalf, col = lane&15 (guard ragged last block)
  const int col0 = n0 + rrow;
  const float bb0 = bias[col0];
  const float bb1 = bias[col0 + 16];
#pragma unroll
  for (int t = 0; t < 4; ++t) {
    const int rbase = brow + t * 16 + khalf * 8;
#pragma unroll
    for (int r = 0; r < 8; ++r) {
      const int row = rbase + r;
      if (row < N) {
        float v0 = acc[t][0][r] + bb0;
        float v1 = acc[t][1][r] + bb1;
        if (RELU) { v0 = fmaxf(v0, 0.0f); v1 = fmaxf(v1, 0.0f); }
        out[(size_t)row * 256 + col0]      = f2b(v0);
        out[(size_t)row * 256 + col0 + 16] = f2b(v1);
      }
    }
  }
}

// ---------- final 256 -> 2 projection (fp32 out) ----------
__global__ void k_final(const unsigned short* __restrict__ h, const float* __restrict__ W4,
                        const float* __restrict__ b4, float* __restrict__ out, int N) {
  int n = blockIdx.x * 256 + threadIdx.x;
  if (n >= N) return;
  float a0 = b4[0], a1 = b4[1];
  const U4* hp = (const U4*)h + (size_t)n * 32;
#pragma unroll 4
  for (int j = 0; j < 32; ++j) {
    U4 d = hp[j];
    unsigned int ws[4] = {d.x, d.y, d.z, d.w};
    int k = j * 8;
#pragma unroll
    for (int i = 0; i < 4; ++i) {
      float lo = b2f(ws[i] & 0xffffu);
      float hi = __uint_as_float(ws[i] & 0xffff0000u);
      a0 += lo * W4[k + 2 * i] + hi * W4[k + 2 * i + 1];
      a1 += lo * W4[256 + k + 2 * i] + hi * W4[256 + k + 2 * i + 1];
    }
  }
  out[(size_t)n * 2]     = a0;
  out[(size_t)n * 2 + 1] = a1;
}

// ---------- launch ----------
extern "C" void kernel_launch(void* const* d_in, const int* in_sizes, int n_in,
                              void* d_out, int out_size, void* d_ws, size_t ws_size,
                              hipStream_t stream) {
  const float* x   = (const float*)d_in[0];
  const int*   ei  = (const int*)d_in[1];
  const float* W1  = (const float*)d_in[2];
  const float* b1  = (const float*)d_in[3];
  const float* W2  = (const float*)d_in[4];
  const float* b2  = (const float*)d_in[5];
  const float* W3  = (const float*)d_in[6];
  const float* b3  = (const float*)d_in[7];
  const float* W4  = (const float*)d_in[8];
  const float* b4  = (const float*)d_in[9];
  const float* T10 = (const float*)d_in[10];
  const float* T11 = (const float*)d_in[11];
  const float* cb1 = (const float*)d_in[12];
  const float* T20 = (const float*)d_in[13];
  const float* T21 = (const float*)d_in[14];
  const float* cb2 = (const float*)d_in[15];
  float* out = (float*)d_out;

  const int N = in_sizes[0] / 256;   // 50000
  const int E = in_sizes[1] / 2;     // 1600000

  // workspace carve-out
  char* p = (char*)d_ws;
  auto alloc = [&](size_t sz) { char* r = p; p += (sz + 511) & ~(size_t)511; return r; };
  const size_t act = (size_t)N * 256 * 2;
  unsigned short* bufA = (unsigned short*)alloc(act);          // xb -> h3
  unsigned short* bufB = (unsigned short*)alloc(act);          // h1 -> h4
  unsigned short* bufC = (unsigned short*)alloc(act);          // h2 -> h5
  float*          txf  = (float*)alloc((size_t)N * 256 * 4);   // scatter accumulator
  unsigned short* txb  = (unsigned short*)alloc(act);
  unsigned short* wb[7];                                        // W1 W2 W3 T10 T11 T20 T21
  for (int i = 0; i < 7; ++i) wb[i] = (unsigned short*)alloc((size_t)256 * 256 * 2);
  float* deg = (float*)alloc((size_t)N * 4);
  float* dis = (float*)alloc((size_t)N * 4);

  const int TB = 256;
  auto cdiv = [](int a, int b) { return (a + b - 1) / b; };
  const int nF    = N * 256;
  const int nW    = 256 * 256;
  const int gGemm = cdiv(N, 64);      // 782 blocks of 64 rows

  // degree -> dis
  k_zero<<<cdiv(N, TB), TB, 0, stream>>>(deg, N);
  k_deg<<<cdiv(E, TB), TB, 0, stream>>>(ei, deg, E);
  k_rsqrt<<<cdiv(N, TB), TB, 0, stream>>>(deg, dis, N);

  // bf16 conversions
  k_cvt<<<cdiv(nF, TB), TB, 0, stream>>>(x, bufA, nF);
  const float* wsrc[7] = {W1, W2, W3, T10, T11, T20, T21};
  for (int i = 0; i < 7; ++i)
    k_cvt<<<cdiv(nW, TB), TB, 0, stream>>>(wsrc[i], wb[i], nW);

  // h1 = relu(x@W1^T+b1), h2 = relu(h1@W2^T+b2)
  k_gemm<true, false><<<gGemm, TB, 0, stream>>>(bufA, wb[0], bufA, wb[0], b1, bufB, N);
  k_gemm<true, false><<<gGemm, TB, 0, stream>>>(bufB, wb[1], bufB, wb[1], b2, bufC, N);

  // cheb 1: tx1 = segsum(norm*h2[src] -> dst);  h3 = h2@T10^T + tx1@T11^T + cb1
  k_zero<<<cdiv(nF, TB), TB, 0, stream>>>(txf, nF);
  k_scatter<<<E / 8, TB, 0, stream>>>(ei, dis, bufC, txf, E);
  k_cvt<<<cdiv(nF, TB), TB, 0, stream>>>(txf, txb, nF);
  k_gemm<false, true><<<gGemm, TB, 0, stream>>>(bufC, wb[3], txb, wb[4], cb1, bufA, N);

  // cheb 2
  k_zero<<<cdiv(nF, TB), TB, 0, stream>>>(txf, nF);
  k_scatter<<<E / 8, TB, 0, stream>>>(ei, dis, bufA, txf, E);
  k_cvt<<<cdiv(nF, TB), TB, 0, stream>>>(txf, txb, nF);
  k_gemm<false, true><<<gGemm, TB, 0, stream>>>(bufA, wb[5], txb, wb[6], cb2, bufB, N);

  // h5 = relu(h4@W3^T+b3); out = h5@W4^T + b4
  k_gemm<true, false><<<gGemm, TB, 0, stream>>>(bufB, wb[2], bufB, wb[2], b3, bufC, N);
  k_final<<<cdiv(N, TB), TB, 0, stream>>>(bufC, W4, b4, out, N);
}
